// RWKV6TimeMix_41815801594116
// MI455X (gfx1250) — matrix operand order repaired and checked by reading, not compiled, not run on hardware
//
#include <hip/hip_runtime.h>
#include <hip/hip_bf16.h>
#include <math.h>

typedef __attribute__((ext_vector_type(16))) __bf16 v16bf;
typedef __attribute__((ext_vector_type(8)))  float  v8f;

union FragV {
    v16bf v;
    uint4 q[2];          // two 16-byte halves = 8+8 packed bf16
};

__device__ __forceinline__ unsigned short f2bf(float f) {
    unsigned u = __float_as_uint(f);
    unsigned r = u + 0x7FFFu + ((u >> 16) & 1u);   // round-to-nearest-even
    return (unsigned short)(r >> 16);
}

// ---------------------------------------------------------------------------
// fp32 -> packed bf16 conversion (one pass, vectorized 4-wide)
// ---------------------------------------------------------------------------
__global__ void cvt_f32_to_bf16(const float* __restrict__ in,
                                unsigned short* __restrict__ out, long n4) {
    const long i = blockIdx.x * (long)blockDim.x + threadIdx.x;
    if (i >= n4) return;
    const float4 f = *reinterpret_cast<const float4*>(in + i * 4);
    const unsigned lo = (unsigned)f2bf(f.x) | ((unsigned)f2bf(f.y) << 16);
    const unsigned hi = (unsigned)f2bf(f.z) | ((unsigned)f2bf(f.w) << 16);
    *reinterpret_cast<uint2*>(out + i * 4) = make_uint2(lo, hi);
}

// ---------------------------------------------------------------------------
// C[M,N] = A[M,K] @ B[N,K]^T, packed bf16 operands, fp32 accumulate.
// K is a compile-time constant -> all fragment loads become
// base + 24-bit immediate offsets (no per-row 64-bit address registers).
// Per wave: 16(M) x 64(N) tile; double-buffered K-loop (2 x 32 per iter).
// Requires K % 64 == 0.
// EPI: 0 none, 1 sigmoid, 2 tanh, 3 -softplus(bias[n]+v),
//      4 sigmoid only for nOut >= N-1024 (fused r/k/v/g epilogue)
// OUT_BF16: write C as packed bf16 (feeds a following GEMM) or fp32.
// ---------------------------------------------------------------------------
template <int EPI, int OUT_BF16, int K>
__global__ __launch_bounds__(256)
void gemm_bf16(const unsigned short* __restrict__ A,
               const unsigned short* __restrict__ B,
               void* __restrict__ Cv,
               int M, int N,
               const float* __restrict__ bias) {
    const int lane   = threadIdx.x & 31;
    const int wave   = threadIdx.x >> 5;
    const int nBase  = (blockIdx.x * 8 + wave) * 64;   // 4 n-subtiles per wave
    if (nBase >= N) return;
    const int mTile  = blockIdx.y;

    const int laneLo = lane & 15;
    const int laneHi = lane >> 4;
    const int kHalfA = laneHi * 8;     // A frag: {K0..7,K16..23} / {K8..15,K24..31}
    const int kHalfB = laneHi * 8;     // B frag: {K0..7,K16..23} / {K8..15,K24..31}

    const unsigned short* aRow  = A + (long)(mTile * 16 + laneLo) * K + kHalfA;
    const unsigned short* bRow0 = B + (long)(nBase + laneLo) * K + kHalfB;

    v8f acc[4] = {};

    auto loadFrag = [&](FragV& a, FragV* b, int kb) {
        a.q[0] = *reinterpret_cast<const uint4*>(aRow + kb);
        a.q[1] = *reinterpret_cast<const uint4*>(aRow + kb + 16);
        #pragma unroll
        for (int s = 0; s < 4; ++s) {
            const unsigned short* p = bRow0 + s * 16 * K + kb;  // s*16*K: constant
            b[s].q[0] = *reinterpret_cast<const uint4*>(p);
            b[s].q[1] = *reinterpret_cast<const uint4*>(p + 16);
        }
    };
    auto mma4 = [&](const FragV& a, const FragV* b) {
        #pragma unroll
        for (int s = 0; s < 4; ++s)
            acc[s] = __builtin_amdgcn_wmma_f32_16x16x32_bf16(
                false, a.v, false, b[s].v, (short)0, acc[s], false, false);
    };

    FragV a0, a1, b0[4], b1[4];
    loadFrag(a0, b0, 0);
    loadFrag(a1, b1, 32);
    for (int kb = 0; kb + 64 < K; kb += 64) {
        __builtin_prefetch(aRow + kb + 256, 0, 1);
        mma4(a0, b0);
        loadFrag(a0, b0, kb + 64);      // consumed next iteration
        mma4(a1, b1);
        loadFrag(a1, b1, kb + 96);
    }
    mma4(a0, b0);
    mma4(a1, b1);

    const int mBase = mTile * 16 + laneHi * 8;
    #pragma unroll
    for (int s = 0; s < 4; ++s) {
        const int nOut = nBase + s * 16 + laneLo;
        #pragma unroll
        for (int r = 0; r < 8; ++r) {
            float v = acc[s][r];
            if (EPI == 1) {
                v = 1.f / (1.f + __expf(-v));
            } else if (EPI == 2) {
                v = tanhf(v);
            } else if (EPI == 3) {
                const float wv = bias[nOut] + v;
                const float sp = (wv > 20.f) ? wv : log1pf(__expf(wv));
                v = -sp;
            } else if (EPI == 4) {
                if (nOut >= N - 1024) v = 1.f / (1.f + __expf(-v));
            }
            const long ci = (long)(mBase + r) * N + nOut;
            if (OUT_BF16) ((unsigned short*)Cv)[ci] = f2bf(v);
            else          ((float*)Cv)[ci] = v;
        }
    }
}

// ---------------------------------------------------------------------------
// WKV6 scan: one block per (b,h); 64 threads, thread j owns state column j
// (64 fp32 in VGPRs). r/k/v/exp(w) broadcast via LDS each timestep.
// r/k/v read with row stride ldR (fused rkvg buffer); w with ldW; out ldO.
// ---------------------------------------------------------------------------
__global__ void wkv6_scan(const float* __restrict__ R_,
                          const float* __restrict__ K_,
                          const float* __restrict__ V_, int ldR,
                          const float* __restrict__ W_, int ldW,
                          const float* __restrict__ U_,
                          float* __restrict__ O_, int ldO) {
    const int bh = blockIdx.x;           // 0..63
    const int b  = bh >> 4;
    const int h  = bh & 15;
    const int j  = threadIdx.x;          // 0..63

    __shared__ float rs[64], ks[64], vs[64], ew[64], us[64];
    us[j] = U_[h * 64 + j];

    float state[64];
    #pragma unroll
    for (int i = 0; i < 64; ++i) state[i] = 0.f;

    const int hd = h * 64 + j;
    for (int t = 0; t < 1024; ++t) {
        __syncthreads();
        const long row = (long)b * 1024 + t;
        rs[j] = R_[row * ldR + hd];
        ks[j] = K_[row * ldR + hd];
        vs[j] = V_[row * ldR + hd];
        ew[j] = __expf(W_[row * ldW + hd]);
        __syncthreads();

        const float vj = vs[j];
        float acc = 0.f;
        #pragma unroll
        for (int i = 0; i < 64; ++i) {
            const float kv = ks[i] * vj;
            acc += rs[i] * (state[i] + us[i] * kv);
            state[i] = state[i] * ew[i] + kv;
        }
        O_[row * ldO + hd] = acc;
    }
}

// ---------------------------------------------------------------------------
// Per-head LayerNorm (Dh=64) * sigmoid gate; output packed bf16 (feeds o-GEMM)
// ---------------------------------------------------------------------------
__global__ void ln_gate(const float* __restrict__ oh,
                        const float* __restrict__ g, int ldG,
                        const float* __restrict__ gamma,
                        const float* __restrict__ beta,
                        unsigned short* __restrict__ y,
                        int nRowHead) {
    const int t = blockIdx.x * blockDim.x + threadIdx.x;
    if (t >= nRowHead) return;
    const int row = t >> 4;
    const int h   = t & 15;
    const long base  = (long)row * 1024 + h * 64;
    const long baseG = (long)row * ldG + h * 64;

    float mu = 0.f;
    #pragma unroll 8
    for (int i = 0; i < 64; ++i) mu += oh[base + i];
    mu *= (1.f / 64.f);
    float var = 0.f;
    #pragma unroll 8
    for (int i = 0; i < 64; ++i) {
        const float d = oh[base + i] - mu;
        var += d * d;
    }
    var *= (1.f / 64.f);
    const float rstd = rsqrtf(var + 1e-5f);
    #pragma unroll 8
    for (int i = 0; i < 64; ++i) {
        const float n = (oh[base + i] - mu) * rstd * gamma[i] + beta[i];
        y[base + i] = f2bf(n * g[baseG + i]);
    }
}

// ---------------------------------------------------------------------------
extern "C" void kernel_launch(void* const* d_in, const int* in_sizes, int n_in,
                              void* d_out, int out_size, void* d_ws, size_t ws_size,
                              hipStream_t stream) {
    (void)in_sizes; (void)n_in; (void)out_size; (void)ws_size;

    const int Bb = 4, T = 1024, Dm = 1024, H = 16, Rr = 64;
    const int M = Bb * T;                 // 4096 rows

    const float* x      = (const float*)d_in[0];
    const float* w_r    = (const float*)d_in[1];
    const float* w_k    = (const float*)d_in[2];
    const float* w_v    = (const float*)d_in[3];
    const float* w_o    = (const float*)d_in[4];
    const float* w_g    = (const float*)d_in[5];
    const float* w_base = (const float*)d_in[6];
    const float* w_ld   = (const float*)d_in[7];   // (64, 1024)
    const float* w_lu   = (const float*)d_in[8];   // (1024, 64)
    const float* u      = (const float*)d_in[9];
    const float* gamma  = (const float*)d_in[10];
    const float* beta   = (const float*)d_in[11];
    float*       out    = (float*)d_out;

    // ---- workspace carve-up (256B aligned) ----
    char* wsp = (char*)d_ws;
    auto carve = [&](size_t bytes) {
        char* p = wsp;
        wsp += (bytes + 255) & ~(size_t)255;
        return p;
    };
    unsigned short* xb   = (unsigned short*)carve((size_t)M * Dm * 2);
    unsigned short* wcat = (unsigned short*)carve((size_t)4 * Dm * Dm * 2); // r|k|v|g
    unsigned short* wob  = (unsigned short*)carve((size_t)Dm * Dm * 2);
    unsigned short* wldb = (unsigned short*)carve((size_t)Rr * Dm * 2);
    unsigned short* wlub = (unsigned short*)carve((size_t)Dm * Rr * 2);
    float* rkvg = (float*)carve((size_t)M * 4 * Dm * 4);   // M x 4096 fused
    float* w_   = (float*)carve((size_t)M * Dm * 4);
    unsigned short* h_ = (unsigned short*)carve((size_t)M * Rr * 2);
    float* oh   = (float*)carve((size_t)M * Dm * 4);
    unsigned short* yb = xb;              // xb dead after projection GEMMs

    // ---- 1) one-shot bf16 conversion of activations + weights ----
    const dim3 cb(256);
    auto cvt = [&](const float* src, unsigned short* dst, long n) {
        const long n4 = n / 4;
        cvt_f32_to_bf16<<<dim3((unsigned)((n4 + 255) / 256)), cb, 0, stream>>>(src, dst, n4);
    };
    cvt(x,    xb,                     (long)M * Dm);
    cvt(w_r,  wcat + 0L * Dm * Dm,    (long)Dm * Dm);
    cvt(w_k,  wcat + 1L * Dm * Dm,    (long)Dm * Dm);
    cvt(w_v,  wcat + 2L * Dm * Dm,    (long)Dm * Dm);
    cvt(w_g,  wcat + 3L * Dm * Dm,    (long)Dm * Dm);
    cvt(w_o,  wob,                    (long)Dm * Dm);
    cvt(w_ld, wldb,                   (long)Rr * Dm);
    cvt(w_lu, wlub,                   (long)Dm * Rr);

    // ---- 2) projection GEMMs (WMMA bf16, fp32 acc) ----
    const dim3 blk(256);
    const dim3 grdRKVG((4 * Dm + 511) / 512, M / 16);   // N=4096 -> grid.x = 8
    const dim3 grdFull((Dm + 511) / 512, M / 16);       // N=1024 -> grid.x = 2
    const dim3 grdLora((Rr + 511) / 512, M / 16);       // N=64   -> grid.x = 1

    // fused r|k|v|g projection; sigmoid on the gate quarter (last 1024 cols)
    gemm_bf16<4, 0, 1024><<<grdRKVG, blk, 0, stream>>>(xb, wcat, rkvg, M, 4 * Dm, nullptr);
    // LoRA: h = tanh(x @ w_ld^T) in bf16, then w = -softplus(w_base + h @ w_lu^T)
    gemm_bf16<2, 1, 1024><<<grdLora, blk, 0, stream>>>(xb, wldb, h_, M, Rr, nullptr);
    gemm_bf16<3, 0, 64><<<grdFull, blk, 0, stream>>>(h_, wlub, w_, M, Dm, w_base);

    // ---- 3) sequential WKV6 scan (B*H = 64 independent state machines) ----
    wkv6_scan<<<Bb * H, 64, 0, stream>>>(rkvg + 0 * Dm, rkvg + 1 * Dm, rkvg + 2 * Dm,
                                         4 * Dm, w_, Dm, u, oh, Dm);

    // ---- 4) LayerNorm * gate -> bf16 ----
    ln_gate<<<(M * H + 255) / 256, 256, 0, stream>>>(oh, rkvg + 3 * Dm, 4 * Dm,
                                                     gamma, beta, yb, M * H);

    // ---- 5) output projection ----
    gemm_bf16<0, 0, 1024><<<grdFull, blk, 0, stream>>>(yb, wob, out, M, Dm, nullptr);
}
